// GAT_SAGPool_5944234737697
// MI455X (gfx1250) — compile-verified, matrix operand order checked by reading, not run on hardware
//
#include <hip/hip_runtime.h>
#include <hip/hip_bf16.h>
#include <math.h>

// ---------------- problem constants (from reference) ----------------
#define BGR   4          // graphs
#define NG0   2500       // nodes per graph at entry
#define ETOT0 100000     // original edges
#define HH    2          // heads
#define CC    512        // channels
#define HC    1024       // H*C
#define NEGS  0.2f
__constant__ const int KS_[4] = {1750, 1225, 858, 601};

typedef __bf16 bf16_t;
typedef __attribute__((ext_vector_type(16))) __bf16 v16bf;
typedef __attribute__((ext_vector_type(8)))  float  v8f;

// ---------------- helpers ----------------
static __device__ __forceinline__ bf16_t f2bf(float f) {
  unsigned u = __float_as_uint(f);
  unsigned r = (u + 0x7FFFu + ((u >> 16) & 1u)) >> 16;   // RNE
  unsigned short s = (unsigned short)r;
  return __builtin_bit_cast(bf16_t, s);
}
static __device__ __forceinline__ unsigned fenc(float f) {  // order-preserving
  unsigned u = __float_as_uint(f);
  return (u & 0x80000000u) ? ~u : (u | 0x80000000u);
}
static __device__ __forceinline__ float fdec(unsigned e) {
  return (e & 0x80000000u) ? __uint_as_float(e & 0x7FFFFFFFu)
                           : __uint_as_float(~e);
}

// ---------------- tiny utility kernels ----------------
__global__ void k_zero_f32(float* p, int n) {
  int t = blockIdx.x * blockDim.x + threadIdx.x;
  if (t < n) p[t] = 0.f;
}
__global__ void k_zero_u32(unsigned* p, int n) {
  int t = blockIdx.x * blockDim.x + threadIdx.x;
  if (t < n) p[t] = 0u;
}
__global__ void k_copy_edges(const int* __restrict__ ei, int* __restrict__ src,
                             int* __restrict__ dst, int* __restrict__ em) {
  int t = blockIdx.x * blockDim.x + threadIdx.x;
  if (t < ETOT0) { src[t] = ei[t]; dst[t] = ei[ETOT0 + t]; em[t] = 1; }
}
__global__ void k_cast_pad(const float* __restrict__ x, bf16_t* __restrict__ o,
                           int valid, int total) {
  int t = blockIdx.x * blockDim.x + threadIdx.x;
  if (t < total) o[t] = (t < valid) ? f2bf(x[t]) : __builtin_bit_cast(bf16_t, (unsigned short)0);
}
// W [512,1024] row-major f32 -> Wt [1024,512] n-major bf16
__global__ void k_transpose_w(const float* __restrict__ W, bf16_t* __restrict__ Wt) {
  int t = blockIdx.x * blockDim.x + threadIdx.x;
  if (t < 512 * 1024) {
    int kk = t / 1024, nn = t - kk * 1024;
    Wt[(size_t)nn * 512 + kk] = f2bf(W[t]);
  }
}

// ---------------- WMMA bf16 GEMM:  Out[M,1024] = A[M,512] * W + bias ------
// one wave -> 32(M) x 64(N) tile; 2x4 v8f accumulators; 16 K-steps of 32
__global__ __launch_bounds__(128) void k_gemm_bf16(
    const bf16_t* __restrict__ A, const bf16_t* __restrict__ Bt,
    const float* __restrict__ bias, float* __restrict__ Out,
    int Mvalid, int Mpad) {
  const int lane = threadIdx.x & 31;
  const int wid  = blockIdx.x * (blockDim.x >> 5) + (threadIdx.x >> 5);
  const int tm = wid >> 4;          // 16 N-tiles of width 64
  const int tn = wid & 15;
  const int m0 = tm * 32;
  if (m0 >= Mpad) return;
  const int n0   = tn * 64;
  const int half = lane >> 4;       // K half (0 -> K0..15, 1 -> K16..31)
  const int l16  = lane & 15;

  const bf16_t* a0p = A  + (size_t)(m0 + l16) * 512 + half * 16;
  const bf16_t* a1p = a0p + (size_t)16 * 512;
  const bf16_t* bp  = Bt + (size_t)(n0 + l16) * 512 + half * 16;

  v8f acc[2][4];
#pragma unroll
  for (int i = 0; i < 2; ++i)
#pragma unroll
    for (int j = 0; j < 4; ++j) acc[i][j] = (v8f){0,0,0,0,0,0,0,0};

  for (int kk = 0; kk < 512; kk += 32) {
    __builtin_prefetch(a0p + kk + 128, 0, 0);
    __builtin_prefetch(bp  + kk + 128, 0, 0);
    v16bf a0 = *(const v16bf*)(a0p + kk);
    v16bf a1 = *(const v16bf*)(a1p + kk);
#pragma unroll
    for (int j = 0; j < 4; ++j) {
      v16bf b = *(const v16bf*)(bp + (size_t)j * 16 * 512 + kk);
      acc[0][j] = __builtin_amdgcn_wmma_f32_16x16x32_bf16(
          false, a0, false, b, (short)0, acc[0][j], false, false);
      acc[1][j] = __builtin_amdgcn_wmma_f32_16x16x32_bf16(
          false, a1, false, b, (short)0, acc[1][j], false, false);
    }
  }
  const int nb = n0 + l16;
#pragma unroll
  for (int i = 0; i < 2; ++i)
#pragma unroll
    for (int r = 0; r < 8; ++r) {
      int m = m0 + i * 16 + r + 8 * half;
      if (m < Mvalid) {
#pragma unroll
        for (int j = 0; j < 4; ++j) {
          int n = nb + j * 16;
          Out[(size_t)m * HC + n] = acc[i][j][r] + bias[n];
        }
      }
    }
}

// ---------------- GATv2 edge kernels ----------------
// one wave per edge (incl. self loops appended after E0)
__global__ __launch_bounds__(256) void k_edge_logits(
    const float* __restrict__ xl, const float* __restrict__ xr,
    const int* __restrict__ src, const int* __restrict__ dst,
    const int* __restrict__ em, const float* __restrict__ att,
    float* __restrict__ logit, int E0, int N) {
  int wid = blockIdx.x * 8 + (threadIdx.x >> 5);
  int lane = threadIdx.x & 31;
  int Etot = E0 + N;
  if (wid >= Etot) return;
  int s, d, m;
  if (wid < E0) { s = src[wid]; d = dst[wid]; m = em[wid]; }
  else          { s = wid - E0; d = s; m = 1; }
#pragma unroll
  for (int h = 0; h < HH; ++h) {
    const float* pl = xl + (size_t)s * HC + h * CC + lane * 16;
    const float* pr = xr + (size_t)d * HC + h * CC + lane * 16;
    const float* pa = att + h * CC + lane * 16;
    float acc = 0.f;
#pragma unroll
    for (int i = 0; i < 16; ++i) {
      float e = pl[i] + pr[i];
      e = (e > 0.f) ? e : NEGS * e;
      acc += pa[i] * e;
    }
#pragma unroll
    for (int off = 16; off > 0; off >>= 1) acc += __shfl_down(acc, off, 32);
    if (lane == 0) logit[(size_t)wid * HH + h] = m ? acc : -1e9f;
  }
}

__global__ void k_seg_max(const float* __restrict__ logit, const int* __restrict__ dst,
                          unsigned* __restrict__ mxu, int E0, int N) {
  int t = blockIdx.x * blockDim.x + threadIdx.x;
  int Etot = E0 + N;
  if (t >= Etot * HH) return;
  int e = t >> 1, h = t & 1;
  int d = (e < E0) ? dst[e] : (e - E0);
  atomicMax(&mxu[d * HH + h], fenc(logit[t]));
}

__global__ void k_edge_p(const float* __restrict__ logit, const int* __restrict__ dst,
                         const int* __restrict__ em, const unsigned* __restrict__ mxu,
                         float* __restrict__ pbuf, float* __restrict__ den,
                         int E0, int N) {
  int t = blockIdx.x * blockDim.x + threadIdx.x;
  int Etot = E0 + N;
  if (t >= Etot * HH) return;
  int e = t >> 1, h = t & 1;
  int d, m;
  if (e < E0) { d = dst[e]; m = em[e]; } else { d = e - E0; m = 1; }
  float p = 0.f;
  if (m) p = expf(logit[t] - fdec(mxu[d * HH + h]));
  pbuf[t] = p;
  if (p != 0.f) atomicAdd(&den[d * HH + h], p);
}

// one wave per (edge,head): out[d,h,:] += alpha * xl[s,h,:]
__global__ __launch_bounds__(256) void k_edge_agg(
    const float* __restrict__ xl, const float* __restrict__ pbuf,
    const float* __restrict__ den, const int* __restrict__ src,
    const int* __restrict__ dst, float* __restrict__ outp, int E0, int N) {
  int wid = blockIdx.x * 8 + (threadIdx.x >> 5);
  int lane = threadIdx.x & 31;
  int Etot = E0 + N;
  if (wid >= Etot * HH) return;
  int e = wid >> 1, h = wid & 1;
  float p = pbuf[wid];
  if (p == 0.f) return;
  int s, d;
  if (e < E0) { s = src[e]; d = dst[e]; } else { s = e - E0; d = s; }
  float alpha = p / fmaxf(den[d * HH + h], 1e-16f);
  const float* xs = xl   + (size_t)s * HC + h * CC + lane * 16;
  float*       od = outp + (size_t)d * HC + h * CC + lane * 16;
#pragma unroll
  for (int i = 0; i < 16; ++i) atomicAdd(&od[i], xs[i] * alpha);
}

__global__ void k_gat_final(const float* __restrict__ agg, const float* __restrict__ bias,
                            float* __restrict__ xo, int N) {
  int t = blockIdx.x * blockDim.x + threadIdx.x;
  if (t >= N * CC) return;
  int n = t >> 9, c = t & (CC - 1);
  float v = 0.5f * (agg[(size_t)n * HC + c] + agg[(size_t)n * HC + CC + c]) + bias[c];
  xo[t] = (v > 0.f) ? v : 0.f;
}

// ---------------- SAGPool ----------------
__global__ __launch_bounds__(256) void k_sag_scatter(
    const float* __restrict__ x, const int* __restrict__ src,
    const int* __restrict__ dst, const int* __restrict__ em,
    float* __restrict__ agg, int E0) {
  int wid = blockIdx.x * 8 + (threadIdx.x >> 5);
  int lane = threadIdx.x & 31;
  if (wid >= E0 || !em[wid]) return;
  const float* xs = x + (size_t)src[wid] * CC + lane * 16;
  float* ad = agg + (size_t)dst[wid] * CC + lane * 16;
#pragma unroll
  for (int i = 0; i < 16; ++i) atomicAdd(&ad[i], xs[i]);
}

__global__ __launch_bounds__(256) void k_score(
    const float* __restrict__ agg, const float* __restrict__ x,
    const float* __restrict__ Wrel, const float* __restrict__ brel,
    const float* __restrict__ Wroot, float* __restrict__ score, int N) {
  int wid = blockIdx.x * 8 + (threadIdx.x >> 5);
  int lane = threadIdx.x & 31;
  if (wid >= N) return;
  const float* ag = agg + (size_t)wid * CC + lane * 16;
  const float* xx = x   + (size_t)wid * CC + lane * 16;
  const float* wr = Wrel  + lane * 16;
  const float* wo = Wroot + lane * 16;
  float acc = 0.f;
#pragma unroll
  for (int i = 0; i < 16; ++i) acc += ag[i] * wr[i] + xx[i] * wo[i];
#pragma unroll
  for (int off = 16; off > 0; off >>= 1) acc += __shfl_down(acc, off, 32);
  if (lane == 0) score[wid] = acc + brel[0];
}

__global__ void k_init_newidx(int* p, int n) {
  int t = blockIdx.x * blockDim.x + threadIdx.x;
  if (t < n) p[t] = -1;
}

// per-graph bitonic top-k in LDS (4096 64-bit keys: score desc, index asc)
__global__ __launch_bounds__(1024) void k_topk(
    const float* __restrict__ score, int ng, int k, int* __restrict__ globl,
    float* __restrict__ tanhv, int* __restrict__ newidx) {
  __shared__ unsigned long long s[4096];
  int b = blockIdx.x, tid = threadIdx.x;
  for (int i = tid; i < 4096; i += 1024) {
    unsigned long long key = 0ull;
    if (i < ng)
      key = ((unsigned long long)fenc(score[(size_t)b * ng + i]) << 32) |
            (0xFFFFFFFFu - (unsigned)i);
    s[i] = key;
  }
  __syncthreads();
  for (int ksz = 2; ksz <= 4096; ksz <<= 1)
    for (int j = ksz >> 1; j > 0; j >>= 1) {
      for (int i = tid; i < 4096; i += 1024) {
        int l = i ^ j;
        if (l > i) {
          bool up = ((i & ksz) == 0);      // ascending overall
          unsigned long long a = s[i], c = s[l];
          if (up ? (a > c) : (a < c)) { s[i] = c; s[l] = a; }
        }
      }
      __syncthreads();
    }
  for (int i = tid; i < k; i += 1024) {
    unsigned long long key = s[4095 - i];  // largest first
    unsigned idx = 0xFFFFFFFFu - (unsigned)(key & 0xFFFFFFFFu);
    float sc = fdec((unsigned)(key >> 32));
    int g = b * ng + (int)idx;
    globl[b * k + i] = g;
    tanhv[b * k + i] = tanhf(sc);
    newidx[g] = b * k + i;
  }
}

__global__ void k_gather(const float* __restrict__ x, const int* __restrict__ globl,
                         const float* __restrict__ tanhv, float* __restrict__ xo,
                         int Nnew) {
  int t = blockIdx.x * blockDim.x + threadIdx.x;
  if (t >= Nnew * CC) return;
  int nn = t >> 9, c = t & (CC - 1);
  xo[t] = x[(size_t)globl[nn] * CC + c] * tanhv[nn];
}

__global__ void k_remap(const int* __restrict__ newidx, int* src, int* dst, int* em) {
  int t = blockIdx.x * blockDim.x + threadIdx.x;
  if (t >= ETOT0) return;
  int ns = newidx[src[t]], nd = newidx[dst[t]];
  int nm = em[t] && ns >= 0 && nd >= 0;
  src[t] = nm ? ns : 0;
  dst[t] = nm ? nd : 0;
  em[t]  = nm;
}

__global__ void k_readout(const float* __restrict__ x, float* __restrict__ ro, int k) {
  int t = blockIdx.x * blockDim.x + threadIdx.x;
  if (t >= BGR * CC) return;
  int b = t >> 9, c = t & (CC - 1);
  float mx = -INFINITY, sm = 0.f;
  for (int n = 0; n < k; ++n) {
    float v = x[(size_t)(b * k + n) * CC + c];
    mx = fmaxf(mx, v); sm += v;
  }
  ro[b * 2 * CC + c]      += mx;
  ro[b * 2 * CC + CC + c] += sm / (float)k;
}

// ---------------- MLP head ----------------
__global__ void k_mlp1(const float* ro, const float* W1, const float* b1, float* h1) {
  int t = blockIdx.x * blockDim.x + threadIdx.x;
  if (t >= BGR * CC) return;
  int b = t >> 9, j = t & (CC - 1);
  float acc = b1[j];
  for (int i = 0; i < 2 * CC; ++i) acc += ro[b * 2 * CC + i] * W1[(size_t)i * CC + j];
  h1[t] = fmaxf(acc, 0.f);
}
__global__ void k_mlp2(const float* h1, const float* W2, const float* b2, float* h2) {
  int t = blockIdx.x * blockDim.x + threadIdx.x;
  if (t >= BGR * 256) return;
  int b = t >> 8, j = t & 255;
  float acc = b2[j];
  for (int i = 0; i < CC; ++i) acc += h1[b * CC + i] * W2[(size_t)i * 256 + j];
  h2[t] = fmaxf(acc, 0.f);
}
__global__ void k_mlp3(const float* h2, const float* W3, const float* b3, float* out) {
  int t = blockIdx.x * blockDim.x + threadIdx.x;
  if (t >= BGR * 5) return;
  int b = t / 5, j = t % 5;
  float acc = b3[j];
  for (int i = 0; i < 256; ++i) acc += h2[b * 256 + i] * W3[i * 5 + j];
  out[t] = acc;
}
__global__ void k_softmax(float* out) {
  int b = threadIdx.x;
  if (b >= BGR) return;
  float mx = -INFINITY;
  for (int j = 0; j < 5; ++j) mx = fmaxf(mx, out[b * 5 + j]);
  float sm = 0.f;
  for (int j = 0; j < 5; ++j) sm += expf(out[b * 5 + j] - mx);
  for (int j = 0; j < 5; ++j)
    out[BGR * 5 + b * 5 + j] = expf(out[b * 5 + j] - mx) / sm;
}

// ---------------- host orchestration ----------------
static inline int cdiv(long long a, long long b) { return (int)((a + b - 1) / b); }

extern "C" void kernel_launch(void* const* d_in, const int* in_sizes, int n_in,
                              void* d_out, int out_size, void* d_ws, size_t ws_size,
                              hipStream_t stream) {
  (void)in_sizes; (void)n_in; (void)out_size; (void)ws_size;
  const float* x0      = (const float*)d_in[0];
  const int*   ei      = (const int*)  d_in[1];
  const float* gWl     = (const float*)d_in[2];
  const float* gbl     = (const float*)d_in[3];
  const float* gWr     = (const float*)d_in[4];
  const float* gbr     = (const float*)d_in[5];
  const float* gatt    = (const float*)d_in[6];
  const float* gbias   = (const float*)d_in[7];
  const float* sWrel   = (const float*)d_in[8];
  const float* sbrel   = (const float*)d_in[9];
  const float* sWroot  = (const float*)d_in[10];
  const float* W1 = (const float*)d_in[11]; const float* b1 = (const float*)d_in[12];
  const float* W2 = (const float*)d_in[13]; const float* b2 = (const float*)d_in[14];
  const float* W3 = (const float*)d_in[15]; const float* b3 = (const float*)d_in[16];
  float* out = (float*)d_out;

  const int KS[4] = {1750, 1225, 858, 601};
  const int NMAX = (BGR * NG0 + 31) & ~31;   // 10016
  const int EP   = ETOT0 + NMAX;

  char* p = (char*)d_ws;
  auto alloc = [&](size_t bytes) -> void* {
    void* r = (void*)p; p += (bytes + 255) & ~(size_t)255; return r;
  };
  bf16_t* xbf   = (bf16_t*)alloc((size_t)NMAX * CC * 2);
  bf16_t* wbf   = (bf16_t*)alloc((size_t)HC * CC * 2);
  float*  xl    = (float*) alloc((size_t)NMAX * HC * 4);   // also SAG agg (first N*CC)
  float*  xr    = (float*) alloc((size_t)NMAX * HC * 4);   // also GAT head accumulator
  float*  xA    = (float*) alloc((size_t)NMAX * CC * 4);   // GAT output
  float*  xB    = (float*) alloc((size_t)NMAX * CC * 4);   // pooled output
  int*    srcw  = (int*)   alloc((size_t)ETOT0 * 4);
  int*    dstw  = (int*)   alloc((size_t)ETOT0 * 4);
  int*    emw   = (int*)   alloc((size_t)ETOT0 * 4);
  float*  logit = (float*) alloc((size_t)EP * HH * 4);
  float*  pbuf  = (float*) alloc((size_t)EP * HH * 4);
  unsigned* mxu = (unsigned*)alloc((size_t)NMAX * HH * 4);
  float*  den   = (float*) alloc((size_t)NMAX * HH * 4);
  float*  score = (float*) alloc((size_t)NMAX * 4);
  int*    newid = (int*)   alloc((size_t)NMAX * 4);
  int*    globl = (int*)   alloc((size_t)BGR * 1750 * 4);
  float*  tanhv = (float*) alloc((size_t)BGR * 1750 * 4);
  float*  ro    = (float*) alloc((size_t)BGR * 2 * CC * 4);
  float*  h1    = (float*) alloc((size_t)BGR * CC * 4);
  float*  h2    = (float*) alloc((size_t)BGR * 256 * 4);

  k_copy_edges<<<cdiv(ETOT0, 256), 256, 0, stream>>>(ei, srcw, dstw, emw);
  k_zero_f32<<<cdiv(BGR * 2 * CC, 256), 256, 0, stream>>>(ro, BGR * 2 * CC);

  int ng = NG0;
  const float* xp = x0;
  for (int L = 0; L < 4; ++L) {
    const int N = BGR * ng, Mpad = (N + 31) & ~31;
    const int kk = KS[L], Nnew = BGR * kk, Etot = ETOT0 + N;

    k_cast_pad<<<cdiv((long long)Mpad * CC, 256), 256, 0, stream>>>(
        xp, xbf, N * CC, Mpad * CC);

    const int gemmWaves = (Mpad / 32) * 16;
    // xl = x @ Wl + bl
    k_transpose_w<<<cdiv(CC * HC, 256), 256, 0, stream>>>(gWl + (size_t)L * CC * HC, wbf);
    k_gemm_bf16<<<cdiv(gemmWaves, 4), 128, 0, stream>>>(xbf, wbf, gbl + L * HC, xl, N, Mpad);
    // xr = x @ Wr + br
    k_transpose_w<<<cdiv(CC * HC, 256), 256, 0, stream>>>(gWr + (size_t)L * CC * HC, wbf);
    k_gemm_bf16<<<cdiv(gemmWaves, 4), 128, 0, stream>>>(xbf, wbf, gbr + L * HC, xr, N, Mpad);

    k_edge_logits<<<cdiv(Etot, 8), 256, 0, stream>>>(
        xl, xr, srcw, dstw, emw, gatt + (size_t)L * HH * CC, logit, ETOT0, N);
    k_zero_u32<<<cdiv(N * HH, 256), 256, 0, stream>>>(mxu, N * HH);
    k_seg_max<<<cdiv(Etot * HH, 256), 256, 0, stream>>>(logit, dstw, mxu, ETOT0, N);
    k_zero_f32<<<cdiv(N * HH, 256), 256, 0, stream>>>(den, N * HH);
    k_edge_p<<<cdiv(Etot * HH, 256), 256, 0, stream>>>(
        logit, dstw, emw, mxu, pbuf, den, ETOT0, N);
    k_zero_f32<<<cdiv((long long)N * HC, 256), 256, 0, stream>>>(xr, N * HC);  // reuse as accum
    k_edge_agg<<<cdiv(Etot * HH, 8), 256, 0, stream>>>(
        xl, pbuf, den, srcw, dstw, xr, ETOT0, N);
    k_gat_final<<<cdiv((long long)N * CC, 256), 256, 0, stream>>>(
        xr, gbias + L * CC, xA, N);

    // ---- SAGPool ----
    k_zero_f32<<<cdiv((long long)N * CC, 256), 256, 0, stream>>>(xl, N * CC); // reuse as agg
    k_sag_scatter<<<cdiv(ETOT0, 8), 256, 0, stream>>>(xA, srcw, dstw, emw, xl, ETOT0);
    k_score<<<cdiv(N, 8), 256, 0, stream>>>(
        xl, xA, sWrel + L * CC, sbrel + L, sWroot + L * CC, score, N);
    k_init_newidx<<<cdiv(N, 256), 256, 0, stream>>>(newid, N);
    k_topk<<<BGR, 1024, 0, stream>>>(score, ng, kk, globl, tanhv, newid);
    k_gather<<<cdiv((long long)Nnew * CC, 256), 256, 0, stream>>>(xA, globl, tanhv, xB, Nnew);
    k_remap<<<cdiv(ETOT0, 256), 256, 0, stream>>>(newid, srcw, dstw, emw);
    k_readout<<<cdiv(BGR * CC, 256), 256, 0, stream>>>(xB, ro, kk);

    xp = xB;
    ng = kk;
  }

  k_mlp1<<<cdiv(BGR * CC, 256), 256, 0, stream>>>(ro, W1, b1, h1);
  k_mlp2<<<cdiv(BGR * 256, 256), 256, 0, stream>>>(h1, W2, b2, h2);
  k_mlp3<<<1, 32, 0, stream>>>(h2, W3, b3, out);
  k_softmax<<<1, BGR, 0, stream>>>(out);
}
